// HybridAttention_49795850830466
// MI455X (gfx1250) — matrix operand order repaired and checked by reading, not compiled, not run on hardware
//
#include <hip/hip_runtime.h>
#include <hip/hip_bf16.h>
#include <stdint.h>

#define BATCH 8
#define CD    256
#define NHEAD 4
#define HDIM  64
#define HH    112
#define WWID  112
#define NWIN  256
#define TOK   49
#define TPAD  64

typedef __bf16 bf16;
typedef __attribute__((ext_vector_type(16))) __bf16 bf16x16;
typedef __attribute__((ext_vector_type(8)))  float  f32x8;

struct alignas(16) U16 { unsigned long long lo, hi; };

union FragH { bf16x16 v; bf16 h[16]; U16 q[2]; };
union AccF  { f32x8 v; float f[8]; };

__device__ __forceinline__ f32x8 wmma_bf16(const FragH& a, const FragH& b, f32x8 c) {
  return __builtin_amdgcn_wmma_f32_16x16x32_bf16(false, a.v, false, b.v, (short)0, c, false, false);
}

__device__ __forceinline__ void zero_acc(AccF& a) {
#pragma unroll
  for (int j = 0; j < 8; ++j) a.f[j] = 0.f;
}

// A fragment from a bf16 row (row pointer already includes m = lane&15 selection).
// Layout: lanes<16 hold K {0..15}, lanes>=16 hold K {16..31} (matches B).
__device__ __forceinline__ FragH loadA_bf16(const bf16* row, int kbase, int lane) {
  const int klo = (lane >> 4) * 16;
  FragH f;
  f.q[0] = *(const U16*)(row + kbase + klo);
  f.q[1] = *(const U16*)(row + kbase + klo + 8);
  return f;
}

// A fragment (16x32) from bf16 row-major weights, ld = 256.
__device__ __forceinline__ FragH loadA_wbf(const bf16* W, int mbase, int kbase, int lane) {
  return loadA_bf16(W + (size_t)(mbase + (lane & 15)) * CD, kbase, lane);
}

// B fragment (32x16): 16 consecutive K for this lane's column (n = lane&15,
// K offset (lane>>4)*16 folded into caller's pointer).
__device__ __forceinline__ FragH loadB_bf16(const bf16* p) {
  FragH f;
  f.q[0] = *(const U16*)(p);
  f.q[1] = *(const U16*)(p + 8);
  return f;
}

// ---------------------------------------------------------------------------
// Prep: one-shot fp32 -> bf16 conversion of the three weight matrices.
// Layout in wbf (elements): [0): w_in (256x256), [65536): w_qkv (768x256),
// [262144): w_out (256x256).
// ---------------------------------------------------------------------------
__global__ void k_wprep(const float* __restrict__ w_in, const float* __restrict__ wqkv,
                        const float* __restrict__ wout, bf16* __restrict__ wbf) {
  const size_t n1 = CD * CD, n2 = (size_t)3 * CD * CD;
  const size_t total = n1 + n2 + n1;
  for (size_t i = (size_t)blockIdx.x * blockDim.x + threadIdx.x; i < total;
       i += (size_t)gridDim.x * blockDim.x) {
    float v;
    if (i < n1)            v = w_in[i];
    else if (i < n1 + n2)  v = wqkv[i - n1];
    else                   v = wout[i - n1 - n2];
    wbf[i] = (bf16)v;
  }
}

// ---------------------------------------------------------------------------
// Kernel 1: per (b, window): xp = SiLU(BN1(w_in*x)); qkv = w_qkv*xp; RoPE(q,k);
// write Q,K [win][t][hd] and V transposed [win][hd][t] (bf16) to workspace.
// ---------------------------------------------------------------------------
__global__ __launch_bounds__(256) void k_qkv(
    const float* __restrict__ x,
    const float* __restrict__ bn1g, const float* __restrict__ bn1b,
    const float* __restrict__ bn1m, const float* __restrict__ bn1v,
    const bf16* __restrict__ wbf,
    bf16* __restrict__ Qws, bf16* __restrict__ Kws, bf16* __restrict__ Vws)
{
  __shared__ bf16 Xt[TPAD * CD];        // [t][c] input tile
  __shared__ bf16 XP[TPAD * CD];        // [t][c] projected tile after BN+SiLU
  __shared__ float cosT[2 * 7 * 32];    // RoPE LUT: [useW][r-or-c][d]
  __shared__ float sinT[2 * 7 * 32];
  __shared__ float sc1[CD], sh1[CD];    // BN1 scale/shift

  const int wi = blockIdx.x, b = blockIdx.y;
  const int hb = wi >> 4, wb = wi & 15;
  const int tid = threadIdx.x, lane = tid & 31, wave = tid >> 5;
  const bf16* w_in_bf = wbf;
  const bf16* wqkv_bf = wbf + CD * CD;

  // Pre-warm L2 with the bf16 weights (global_prefetch_b8).
  for (int idx = tid; idx < CD; idx += 256) {
    __builtin_prefetch(w_in_bf + (size_t)idx * CD, 0, 0);
    __builtin_prefetch(wqkv_bf + (size_t)idx * 3 * CD, 0, 0);
  }
  // RoPE tables + BN1 coefficients.
  for (int idx = tid; idx < 2 * 7 * 32; idx += 256) {
    const int useW = idx >= 224;
    const int rc = (idx - useW * 224) >> 5, d = idx & 31;
    const float pos = (float)(useW ? (wb * 7 + rc) : (hb * 7 + rc));
    const float ang = pos * __powf(10000.f, -(float)(d & ~1) * (1.f / 32.f));
    sinT[idx] = __sinf(ang);
    cosT[idx] = __cosf(ang);
  }
  if (tid < CD) {
    const float sc = bn1g[tid] * rsqrtf(bn1v[tid] + 1e-5f);
    sc1[tid] = sc;
    sh1[tid] = bn1b[tid] - bn1m[tid] * sc;
  }
  // Stage x tile (window gather + fp32->bf16).
  for (int idx = tid; idx < TPAD * CD; idx += 256) {
    const int c = idx >> 6, t = idx & 63;
    float v = 0.f;
    if (t < TOK) {
      const int h = hb * 7 + t / 7, w = wb * 7 + t % 7;
      v = x[((size_t)(b * CD + c) * HH + h) * WWID + w];
    }
    Xt[t * CD + c] = (bf16)v;
  }
  __syncthreads();

  // GEMM1: 256x64 output, 64 tiles over 8 waves.
  for (int job = wave; job < 64; job += 8) {
    const int mt = job >> 2, nt = job & 3;
    AccF acc; zero_acc(acc);
#pragma unroll
    for (int kb = 0; kb < CD; kb += 32) {
      FragH a  = loadA_wbf(w_in_bf, mt * 16, kb, lane);
      FragH bb = loadB_bf16(&Xt[(nt * 16 + (lane & 15)) * CD + kb + ((lane >> 4) * 16)]);
      acc.v = wmma_bf16(a, bb, acc.v);
    }
    const int t  = nt * 16 + (lane & 15);
    const int ob = mt * 16 + ((lane >> 4) * 8);
    FragH outp;
#pragma unroll
    for (int j = 0; j < 8; ++j) {
      const int o = ob + j;
      float v = acc.f[j] * sc1[o] + sh1[o];
      v = v / (1.f + __expf(-v));          // SiLU
      outp.h[j] = (bf16)v;
    }
    *(U16*)&XP[t * CD + ob] = outp.q[0];
  }
  __syncthreads();

  // GEMM2: 768x64 output; M processed in pairs of 16-row tiles (32 channels) so the
  // RoPE rotate-half partner (d <-> d+16) lives in the adjacent accumulator.
  for (int job = wave; job < 96; job += 8) {
    const int mp = job >> 2, nt = job & 3;
    const int obase = mp * 32;
    AccF c0, c1; zero_acc(c0); zero_acc(c1);
#pragma unroll
    for (int kb = 0; kb < CD; kb += 32) {
      FragH bb = loadB_bf16(&XP[(nt * 16 + (lane & 15)) * CD + kb + ((lane >> 4) * 16)]);
      FragH a0 = loadA_wbf(wqkv_bf, obase,      kb, lane);
      FragH a1 = loadA_wbf(wqkv_bf, obase + 16, kb, lane);
      c0.v = wmma_bf16(a0, bb, c0.v);
      c1.v = wmma_bf16(a1, bb, c1.v);
    }
    const int t = nt * 16 + (lane & 15);
    if (t >= TOK) continue;                 // drop pad tokens
    const int r = t / 7, cc = t - r * 7;
    const int rowb = (lane >> 4) * 8;
    if (obase < 2 * CD) {                   // Q or K -> RoPE
      const int oq   = obase & 255;
      const int nh   = oq >> 6;
      const int useW = (oq & 63) >> 5;      // hd<32 uses h-angles; hd>=32 uses w-angles
      const int tb   = useW * 224 + (useW ? cc : r) * 32;
      bf16* dst = (obase < CD ? Qws : Kws) +
                  ((((size_t)b * NHEAD + nh) * NWIN + wi) * TPAD + t) * HDIM + (oq & 32);
      FragH o0, o1;
#pragma unroll
      for (int j = 0; j < 8; ++j) {
        const int d = rowb + j;             // local rotation index 0..15 (partner d+16)
        const float cs0 = cosT[tb + d],      s0 = sinT[tb + d];
        const float cs1 = cosT[tb + d + 16], s1 = sinT[tb + d + 16];
        const float u = c0.f[j], vv = c1.f[j];
        o0.h[j] = (bf16)(u * cs0 - vv * s0);
        o1.h[j] = (bf16)(vv * cs1 + u * s1);
      }
      *(U16*)(dst + rowb)      = o0.q[0];
      *(U16*)(dst + 16 + rowb) = o1.q[0];
    } else {                                // V -> store transposed [hd][t]
      const int ov  = obase - 2 * CD;
      const int nh  = ov >> 6;
      const int hdb = ov & 63;
      bf16* dst = Vws + (((size_t)b * NHEAD + nh) * NWIN + wi) * (size_t)(HDIM * TPAD);
#pragma unroll
      for (int j = 0; j < 8; ++j) {
        dst[(hdb + rowb + j) * TPAD + t]      = (bf16)c0.f[j];
        dst[(hdb + 16 + rowb + j) * TPAD + t] = (bf16)c1.f[j];
      }
    }
  }
}

// ---------------------------------------------------------------------------
// Kernel 2: KV[k,v] = sum_s (elu(K[s,k])+1) * V[s,v] per (b,head); also k2sum.
// K window staged global->LDS with async-to-LDS copies (ASYNCcnt), then
// elu+transpose out of LDS. Partial results via global f32 atomics.
// ---------------------------------------------------------------------------
__global__ __launch_bounds__(256) void k_kv(
    const bf16* __restrict__ Kws, const bf16* __restrict__ Vws,
    float* __restrict__ KV, float* __restrict__ k2sum)
{
  __shared__ bf16 Kraw[TPAD * HDIM];  // raw K window copy (async landing zone)
  __shared__ bf16 K2t[HDIM * TPAD];   // [hd][t] elu(K)+1, pad-masked to 0
  __shared__ float k2p[HDIM];
  const int bnh = blockIdx.y, chunk = blockIdx.x;
  const int tid = threadIdx.x, lane = tid & 31, wave = tid >> 5;
  if (tid < HDIM) k2p[tid] = 0.f;

  AccF acc0, acc1; zero_acc(acc0); zero_acc(acc1);
  const int mt0 = wave >> 2,       nt0 = wave & 3;
  const int mt1 = (wave + 8) >> 2, nt1 = (wave + 8) & 3;

  for (int wl = 0; wl < NWIN / 8; ++wl) {
    const int wi = chunk * (NWIN / 8) + wl;
    const bf16* kb_ = Kws + ((size_t)bnh * NWIN + wi) * (size_t)(TPAD * HDIM);
    __syncthreads();
    // Async global->LDS copy of the 8 KB K window (2 x b128 per thread).
    for (int idx = tid; idx < (TPAD * HDIM) / 8; idx += 256) {
      const bf16* g = kb_ + idx * 8;
      const unsigned l = (unsigned)(uintptr_t)(&Kraw[idx * 8]);
      asm volatile("global_load_async_to_lds_b128 %0, %1, off"
                   :: "v"(l), "v"(g) : "memory");
    }
    asm volatile("s_wait_asynccnt 0x0" ::: "memory");
    __syncthreads();
    // elu+1, pad mask, transpose, k2sum partials.
    for (int idx = tid; idx < TPAD * HDIM; idx += 256) {
      const int t = idx >> 6, hd = idx & 63;
      float v = 0.f;
      if (t < TOK) {
        const float kv = (float)Kraw[t * HDIM + hd];
        v = kv > 0.f ? kv + 1.f : __expf(kv);
        atomicAdd(&k2p[hd], v);
      }
      K2t[hd * TPAD + t] = (bf16)v;
    }
    __syncthreads();
    const bf16* vb = Vws + ((size_t)bnh * NWIN + wi) * (size_t)(HDIM * TPAD);
#pragma unroll
    for (int kb = 0; kb < TPAD; kb += 32) {
      {
        FragH a  = loadA_bf16(&K2t[(mt0 * 16 + (lane & 15)) * TPAD], kb, lane);
        FragH bb = loadB_bf16(vb + (nt0 * 16 + (lane & 15)) * TPAD + kb + ((lane >> 4) * 16));
        acc0.v = wmma_bf16(a, bb, acc0.v);
      }
      {
        FragH a  = loadA_bf16(&K2t[(mt1 * 16 + (lane & 15)) * TPAD], kb, lane);
        FragH bb = loadB_bf16(vb + (nt1 * 16 + (lane & 15)) * TPAD + kb + ((lane >> 4) * 16));
        acc1.v = wmma_bf16(a, bb, acc1.v);
      }
    }
  }
  __syncthreads();
  if (tid < HDIM) atomicAdd(&k2sum[bnh * HDIM + tid], k2p[tid]);
  {
    const int n = nt0 * 16 + (lane & 15), m0 = mt0 * 16 + ((lane >> 4) * 8);
#pragma unroll
    for (int j = 0; j < 8; ++j)
      atomicAdd(&KV[(size_t)bnh * 4096 + (m0 + j) * 64 + n], acc0.f[j]);
  }
  {
    const int n = nt1 * 16 + (lane & 15), m0 = mt1 * 16 + ((lane >> 4) * 8);
#pragma unroll
    for (int j = 0; j < 8; ++j)
      atomicAdd(&KV[(size_t)bnh * 4096 + (m0 + j) * 64 + n], acc1.f[j]);
  }
}

// ---------------------------------------------------------------------------
// Kernel 3: per (b, window): window attention + linear attention + gate mix
// + output projection + BN2 + residual, fully fused.
// ---------------------------------------------------------------------------
__global__ __launch_bounds__(256) void k_attn_out(
    const float* __restrict__ x,
    const float* __restrict__ bn2g, const float* __restrict__ bn2b,
    const float* __restrict__ bn2m, const float* __restrict__ bn2v,
    const float* __restrict__ gate, const bf16* __restrict__ wbf,
    const bf16* __restrict__ Qws, const bf16* __restrict__ Kws, const bf16* __restrict__ Vws,
    const float* __restrict__ KV, const float* __restrict__ k2sum,
    float* __restrict__ out)
{
  __shared__ float S[TPAD * TPAD];       // 16 KB attention scores
  __shared__ bf16  P[TPAD * TPAD];       //  8 KB softmax probs (bf16)
  __shared__ bf16  Q2buf[TPAD * HDIM];   //  8 KB elu(Q)+1 rows (bf16)
  __shared__ bf16  KVt[HDIM * HDIM];     //  8 KB KV^T for current head
  __shared__ float k2s[NHEAD * HDIM];
  __shared__ float den[TPAD];
  __shared__ bf16  Xm[TPAD * CD];        // 32 KB mixed per-token features
  __shared__ float sc2[CD], sh2[CD];     // BN2 scale/shift

  const int wi = blockIdx.x, b = blockIdx.y;
  const int hb = wi >> 4, wb = wi & 15;
  const int tid = threadIdx.x, lane = tid & 31, wave = tid >> 5;
  const float g = 1.f / (1.f + __expf(-gate[0]));
  const bf16* wout_bf = wbf + (size_t)CD * CD + (size_t)3 * CD * CD;

  if (tid < NHEAD * HDIM) k2s[tid] = k2sum[b * NHEAD * HDIM + tid];
  if (tid < CD) {
    const float sc = bn2g[tid] * rsqrtf(bn2v[tid] + 1e-5f);
    sc2[tid] = sc;
    sh2[tid] = bn2b[tid] - bn2m[tid] * sc;
  }

  for (int nh = 0; nh < NHEAD; ++nh) {
    const size_t base = ((size_t)b * NHEAD + nh) * NWIN + wi;
    const bf16* qb  = Qws + base * (size_t)(TPAD * HDIM);
    const bf16* kb_ = Kws + base * (size_t)(TPAD * HDIM);
    const bf16* vb  = Vws + base * (size_t)(HDIM * TPAD);

    // Stage KV^T for this head (overlaps with S-GEMM; consumed after barrier).
    for (int idx = tid; idx < HDIM * HDIM; idx += 256) {
      const int kk = idx >> 6, vv = idx & 63;
      KVt[vv * 64 + kk] = (bf16)KV[(size_t)(b * NHEAD + nh) * 4096 + idx];
    }

    // S = (Q K^T) * HD^-0.5 : 16 tiles over 8 waves.
#pragma unroll
    for (int i = 0; i < 2; ++i) {
      const int tile = wave + i * 8;
      const int mt = tile >> 2, nt = tile & 3;
      AccF acc; zero_acc(acc);
#pragma unroll
      for (int kk = 0; kk < HDIM; kk += 32) {
        FragH a  = loadA_bf16(qb + (mt * 16 + (lane & 15)) * HDIM, kk, lane);
        FragH bb = loadB_bf16(kb_ + (nt * 16 + (lane & 15)) * HDIM + kk + ((lane >> 4) * 16));
        acc.v = wmma_bf16(a, bb, acc.v);
      }
      const int n = nt * 16 + (lane & 15), m0 = mt * 16 + ((lane >> 4) * 8);
#pragma unroll
      for (int j = 0; j < 8; ++j) S[(m0 + j) * TPAD + n] = acc.f[j] * 0.125f;
    }
    __syncthreads();

    // Threads 0..63: masked softmax row -> P.  Threads 64..127: Q2 row + denominator.
    if (tid < TPAD) {
      const int tq = tid;
      if (tq < TOK) {
        float mx = -1e30f;
        for (int c2 = 0; c2 < TOK; ++c2) mx = fmaxf(mx, S[tq * TPAD + c2]);
        float sum = 0.f;
        for (int c2 = 0; c2 < TOK; ++c2) sum += __expf(S[tq * TPAD + c2] - mx);
        const float inv = 1.f / sum;
        for (int c2 = 0; c2 < TPAD; ++c2)
          P[tq * TPAD + c2] = (c2 < TOK) ? (bf16)(__expf(S[tq * TPAD + c2] - mx) * inv)
                                         : (bf16)0.f;
      } else {
        for (int c2 = 0; c2 < TPAD; ++c2) P[tq * TPAD + c2] = (bf16)0.f;
      }
    } else if (tid < 2 * TPAD) {
      const int t = tid - TPAD;
      const bf16* qrow = qb + t * HDIM;
      float dd = 1e-6f;
      for (int d2 = 0; d2 < HDIM; ++d2) {
        float qv = (float)qrow[d2];
        qv = qv > 0.f ? qv + 1.f : __expf(qv);
        Q2buf[t * HDIM + d2] = (bf16)qv;
        dd += qv * k2s[nh * HDIM + d2];
      }
      den[t] = dd;
    }
    __syncthreads();

    // O = P*V ; num = Q2*KV ; mix -> Xm columns for this head.
#pragma unroll
    for (int i = 0; i < 2; ++i) {
      const int tile = wave + i * 8;
      const int mt = tile >> 2, nt = tile & 3;
      AccF o, nu; zero_acc(o); zero_acc(nu);
#pragma unroll
      for (int kk = 0; kk < HDIM; kk += 32) {
        FragH a  = loadA_bf16(&P[(mt * 16 + (lane & 15)) * TPAD], kk, lane);
        FragH bb = loadB_bf16(vb + (nt * 16 + (lane & 15)) * TPAD + kk + ((lane >> 4) * 16));
        o.v = wmma_bf16(a, bb, o.v);
        FragH a2 = loadA_bf16(&Q2buf[(mt * 16 + (lane & 15)) * HDIM], kk, lane);
        FragH b2 = loadB_bf16(&KVt[(nt * 16 + (lane & 15)) * 64 + kk + ((lane >> 4) * 16)]);
        nu.v = wmma_bf16(a2, b2, nu.v);
      }
      const int hd = nt * 16 + (lane & 15), m0 = mt * 16 + ((lane >> 4) * 8);
#pragma unroll
      for (int j = 0; j < 8; ++j) {
        const int t2 = m0 + j;
        const float mixed = g * o.f[j] + (1.f - g) * (nu.f[j] / den[t2]);
        Xm[t2 * CD + nh * HDIM + hd] = (bf16)mixed;
      }
    }
    __syncthreads();
  }

  // Output projection: y = w_out * Xm^T, + BN2 + residual.
  for (int job = wave; job < 64; job += 8) {
    const int mt = job >> 2, nt = job & 3;
    AccF acc; zero_acc(acc);
#pragma unroll
    for (int kk = 0; kk < CD; kk += 32) {
      FragH a  = loadA_wbf(wout_bf, mt * 16, kk, lane);
      FragH bb = loadB_bf16(&Xm[(nt * 16 + (lane & 15)) * CD + kk + ((lane >> 4) * 16)]);
      acc.v = wmma_bf16(a, bb, acc.v);
    }
    const int t = nt * 16 + (lane & 15);
    if (t < TOK) {
      const int h = hb * 7 + t / 7, w = wb * 7 + t % 7;
      const int m0 = mt * 16 + ((lane >> 4) * 8);
#pragma unroll
      for (int j = 0; j < 8; ++j) {
        const int o = m0 + j;
        const size_t ofs = ((size_t)(b * CD + o) * HH + h) * WWID + w;
        out[ofs] = acc.f[j] * sc2[o] + sh2[o] + x[ofs];
      }
    }
  }
}

__global__ void k_zero(U16* p, size_t n) {
  const size_t stride = (size_t)gridDim.x * blockDim.x;
  U16 z; z.lo = 0; z.hi = 0;
  for (size_t i = (size_t)blockIdx.x * blockDim.x + threadIdx.x; i < n; i += stride) p[i] = z;
}

extern "C" void kernel_launch(void* const* d_in, const int* in_sizes, int n_in,
                              void* d_out, int out_size, void* d_ws, size_t ws_size,
                              hipStream_t stream) {
  const float* x    = (const float*)d_in[0];
  const float* w_in = (const float*)d_in[1];
  const float* bn1g = (const float*)d_in[2];
  const float* bn1b = (const float*)d_in[3];
  const float* bn1m = (const float*)d_in[4];
  const float* bn1v = (const float*)d_in[5];
  const float* wqkv = (const float*)d_in[6];
  const float* wout = (const float*)d_in[7];
  const float* bn2g = (const float*)d_in[8];
  const float* bn2b = (const float*)d_in[9];
  const float* bn2m = (const float*)d_in[10];
  const float* bn2v = (const float*)d_in[11];
  const float* gate = (const float*)d_in[12];

  const size_t SZQ = (size_t)BATCH * NHEAD * NWIN * TPAD * HDIM;  // bf16 elems per tensor
  bf16* Qws = (bf16*)d_ws;
  bf16* Kws = Qws + SZQ;
  bf16* Vws = Kws + SZQ;
  float* KV    = (float*)(Vws + SZQ);
  float* k2sum = KV + (size_t)BATCH * NHEAD * HDIM * HDIM;
  bf16*  wbf   = (bf16*)(k2sum + (size_t)BATCH * NHEAD * HDIM);

  // Zero only the atomically-accumulated KV + k2sum region (532 KB).
  const size_t zero_bytes =
      ((size_t)BATCH * NHEAD * HDIM * HDIM + (size_t)BATCH * NHEAD * HDIM) * sizeof(float);
  k_zero<<<64, 256, 0, stream>>>((U16*)KV, zero_bytes / 16);

  // One-shot weight conversion to bf16 (reused by 2048 blocks from L2).
  k_wprep<<<320, 256, 0, stream>>>(w_in, wqkv, wout, wbf);

  dim3 gA(NWIN, BATCH);
  k_qkv<<<gA, 256, 0, stream>>>(x, bn1g, bn1b, bn1m, bn1v, wbf, Qws, Kws, Vws);

  dim3 gB(8, BATCH * NHEAD);
  k_kv<<<gB, 256, 0, stream>>>(Kws, Vws, KV, k2sum);

  dim3 gC(NWIN, BATCH);
  k_attn_out<<<gC, 256, 0, stream>>>(x, bn2g, bn2b, bn2m, bn2v, gate, wbf,
                                     Qws, Kws, Vws, KV, k2sum, (float*)d_out);

  (void)in_sizes; (void)n_in; (void)out_size; (void)ws_size;
}
